// AttentionStack_3375844295093
// MI455X (gfx1250) — matrix operand order repaired and checked by reading, not compiled, not run on hardware
//
#include <hip/hip_runtime.h>
#include <math.h>
#include <stdint.h>

typedef __attribute__((ext_vector_type(16))) _Float16 v16h;
typedef __attribute__((ext_vector_type(8)))  float    v8f;
typedef __attribute__((ext_vector_type(2)))  float    v2f;
typedef __attribute__((ext_vector_type(4)))  unsigned int v4u;
typedef __attribute__((ext_vector_type(8)))  int      v8i;
typedef __attribute__((ext_vector_type(4)))  int      v4i;

#define SEQ    1024
#define EMB    576
#define NHEAD  16
#define DKH    36
#define NLAYER 6
#define BATCH  2
#define MROWS  (BATCH*SEQ)   // 2048
#define FFD    (4*EMB)       // 2304

// ---------------------------------------------------------------------------
// VALU cross-lane reductions via v_permlane16 (no LDS, no dscnt waits).
// Reduction is within each 16-lane row, matching the WMMA C-fragment layout.
// ---------------------------------------------------------------------------
#if __has_builtin(__builtin_amdgcn_permlane16)
#define PL16(v, s0, s1) __builtin_amdgcn_permlane16((v), (v), (int)(s0), (int)(s1), false, false)
#define HAVE_PL16 1
#elif __has_builtin(__builtin_amdgcn_permlane16_b32)
#define PL16(v, s0, s1) __builtin_amdgcn_permlane16_b32((v), (v), (int)(s0), (int)(s1), false, false)
#define HAVE_PL16 1
#else
#define HAVE_PL16 0
#endif

__device__ inline float red16_max(float v) {
#if HAVE_PL16
  int x = __float_as_int(v);
  v = fmaxf(v, __int_as_float(PL16(x, 0x67452301u, 0xEFCDAB89u))); x = __float_as_int(v);
  v = fmaxf(v, __int_as_float(PL16(x, 0x54761032u, 0xDCFE98BAu))); x = __float_as_int(v);
  v = fmaxf(v, __int_as_float(PL16(x, 0x32107654u, 0xBA98FEDCu))); x = __float_as_int(v);
  v = fmaxf(v, __int_as_float(PL16(x, 0xFEDCBA98u, 0x76543210u)));
  return v;
#else
  for (int m = 1; m < 16; m <<= 1) v = fmaxf(v, __shfl_xor(v, m));
  return v;
#endif
}

__device__ inline float red16_sum(float v) {
#if HAVE_PL16
  int x = __float_as_int(v);
  v += __int_as_float(PL16(x, 0x67452301u, 0xEFCDAB89u)); x = __float_as_int(v);
  v += __int_as_float(PL16(x, 0x54761032u, 0xDCFE98BAu)); x = __float_as_int(v);
  v += __int_as_float(PL16(x, 0x32107654u, 0xBA98FEDCu)); x = __float_as_int(v);
  v += __int_as_float(PL16(x, 0xFEDCBA98u, 0x76543210u));
  return v;
#else
  for (int m = 1; m < 16; m <<= 1) v += __shfl_xor(v, m);
  return v;
#endif
}

__device__ inline float red32_sum(float v) {
  v = red16_sum(v);
#if HAVE_PL16 && __has_builtin(__builtin_amdgcn_permlanex16)
  // identity-mapped exchange with the other 16-lane row
  v += __int_as_float(__builtin_amdgcn_permlanex16(
        __float_as_int(v), __float_as_int(v),
        (int)0x76543210u, (int)0xFEDCBA98u, false, false));
#else
  v += __shfl_xor(v, 16);
#endif
  return v;
}

// ---------------------------------------------------------------------------
// TDM: 2-D tile DMA global -> LDS (raw f32). Builds D# groups per ISA §8.3/8.4.
//  group0: count=1 | lds_addr | global_addr(57b) | type=2
//  group1: data_size=4B, tensor_dim0/1, tile_dim0/1, tensor_dim0_stride
// 6-arg builtin variant (clang-23 / therock headers).
// ---------------------------------------------------------------------------
__device__ inline void tdm_load_2d(unsigned lds_addr, const float* gptr,
                                   unsigned tile_x, unsigned tile_y,
                                   unsigned tensor_x, unsigned tensor_y,
                                   unsigned stride_x) {
  unsigned long long ga = (unsigned long long)(uintptr_t)gptr;
  v4u g0;
  g0.x = 1u;                                    // count=1 (user descriptor)
  g0.y = lds_addr;                              // LDS byte address
  g0.z = (unsigned)(ga & 0xFFFFFFFFu);
  g0.w = (unsigned)((ga >> 32) & 0x01FFFFFFu) | (2u << 30);  // type=2
  v8i g1;
  g1[0] = (int)(2u << 16);                      // data_size = 2 (4 bytes)
  g1[1] = (int)((tensor_x & 0xFFFFu) << 16);    // td0[15:0] -> w1[31:16]
  g1[2] = (int)((tensor_x >> 16) | ((tensor_y & 0xFFFFu) << 16));
  g1[3] = (int)(((tensor_y >> 16) & 0xFFFFu) | (tile_x << 16));
  g1[4] = (int)(tile_y & 0xFFFFu);              // tile_dim1; tile_dim2=0
  g1[5] = (int)stride_x;                        // stride0 low 32
  g1[6] = 0;                                    // stride0 hi16 | stride1 lo16
  g1[7] = 0;                                    // stride1 hi32
  v4i z4 = {0, 0, 0, 0};
  v8i z8 = {0, 0, 0, 0, 0, 0, 0, 0};
  __builtin_amdgcn_tensor_load_to_lds(g0, g1, z4, z4, z8, 0);
}

// ---------------------------------------------------------------------------
// Embed: h[b,s,:] = (s==0 ? sos : x[b,s-1,:]) + pos[s,:]
// ---------------------------------------------------------------------------
__global__ __launch_bounds__(256)
void embed_kernel(const float* __restrict__ x, const float* __restrict__ sos,
                  const float* __restrict__ pe0, const float* __restrict__ pe1,
                  const float* __restrict__ pe2, float* __restrict__ h) {
  int idx = blockIdx.x * 256 + threadIdx.x;
  if (idx >= MROWS * EMB) return;
  int e  = idx % EMB;
  int bs = idx / EMB;
  int s  = bs % SEQ;
  int b  = bs / SEQ;
  float v = (s == 0) ? sos[e] : x[((size_t)b * SEQ + (s - 1)) * EMB + e];
  int t = s >> 8, hh = (s >> 4) & 15, w = s & 15;
  float p;
  if (e < 192)      p = pe0[t  * 192 + e];
  else if (e < 384) p = pe1[hh * 192 + (e - 192)];
  else              p = pe2[w  * 192 + (e - 384)];
  h[idx] = v + p;
}

// ---------------------------------------------------------------------------
// LayerNorm: one wave32 per row of [2048, 576]; 576 = 32*18
// ---------------------------------------------------------------------------
__global__ __launch_bounds__(256)
void ln_kernel(const float* __restrict__ x, const float* __restrict__ s,
               const float* __restrict__ b, float* __restrict__ y) {
  int wave = threadIdx.x >> 5;
  int lane = threadIdx.x & 31;
  int row  = blockIdx.x * 8 + wave;
  const float* xr = x + (size_t)row * EMB;
  float vals[18];
  float sum = 0.f;
#pragma unroll
  for (int i = 0; i < 18; ++i) { vals[i] = xr[lane + 32 * i]; sum += vals[i]; }
  sum = red32_sum(sum);
  float mean = sum * (1.0f / EMB);
  float var = 0.f;
#pragma unroll
  for (int i = 0; i < 18; ++i) { float d = vals[i] - mean; var += d * d; }
  var = red32_sum(var) * (1.0f / EMB);
  float r = rsqrtf(var + 1e-5f);
  float* yr = y + (size_t)row * EMB;
#pragma unroll
  for (int i = 0; i < 18; ++i) {
    int e = lane + 32 * i;
    yr[e] = (vals[i] - mean) * r * s[e] + b[e];
  }
}

// ---------------------------------------------------------------------------
// WMMA GEMM with TDM double-buffered staging.
// Block tile 64x64, 256 thr = 8 waves (2x4), wave computes 32x16.
// Wave0 issues tensor_load_to_lds for A(64x32) and W(32x64) f32 tiles,
// 2 chunks in flight; fragments convert f32->f16 on LDS read.
// ---------------------------------------------------------------------------
template <int GELU, int HASRES, int HASBIAS>
__global__ __launch_bounds__(256)
void gemm_wmma(const float* __restrict__ A, const float* __restrict__ W,
               const float* __restrict__ bias, const float* __restrict__ res,
               float* __restrict__ out, int K, int N) {
  __shared__ float As[2][64][32];   // [buf][m][k]
  __shared__ float Bs[2][32][64];   // [buf][k][n]  (row-major as in memory)
  int tid  = threadIdx.x;
  int lane = tid & 31, wave = tid >> 5;
  int wm = wave & 1, wn = wave >> 1;
  int mBlk = blockIdx.x * 64, nBlk = blockIdx.y * 64;
  int l16  = lane & 15;
  int offA = (lane < 16) ? 0 : 8;
  int offB = (lane < 16) ? 0 : 16;
  int nc   = K >> 5;

  unsigned aLds = (unsigned)(uintptr_t)&As[0][0][0];
  unsigned bLds = (unsigned)(uintptr_t)&Bs[0][0][0];
  const unsigned aBufB = 64 * 32 * 4, bBufB = 32 * 64 * 4;

  if (wave == 0) {
    tdm_load_2d(aLds, A + (size_t)mBlk * K, 32, 64, (unsigned)K, MROWS, (unsigned)K);
    tdm_load_2d(bLds, W + nBlk,             64, 32, (unsigned)N, (unsigned)K, (unsigned)N);
    if (nc > 1) {
      tdm_load_2d(aLds + aBufB, A + (size_t)mBlk * K + 32,   32, 64, (unsigned)K, MROWS, (unsigned)K);
      tdm_load_2d(bLds + bBufB, W + (size_t)32 * N + nBlk,   64, 32, (unsigned)N, (unsigned)K, (unsigned)N);
    }
  }

  v8f acc0 = {}; v8f acc1 = {};
  for (int c = 0; c < nc; ++c) {
    int buf = c & 1;
    if (c + 1 < nc) {
      if (wave == 0) __builtin_amdgcn_s_wait_tensorcnt(2);  // chunk c landed
    } else {
      if (wave == 0) __builtin_amdgcn_s_wait_tensorcnt(0);
    }
    __syncthreads();

    // B fragment: lane = col, K = (i<8 ? i : i+8) + (lane<16 ? 0 : 8)
    v16h bfrag;
#pragma unroll
    for (int i = 0; i < 16; ++i)
      bfrag[i] = (_Float16)Bs[buf][((i < 8) ? i : (i + 8)) + offA][wn * 16 + l16];
    // A fragments: lane = row, K = (i<8 ? i : i+8) + (lane<16 ? 0 : 8)
    v16h af0, af1;
#pragma unroll
    for (int j = 0; j < 8; ++j) {
      int kk = ((j < 4) ? (2 * j) : (2 * j + 8)) + offA;
      v2f p0 = *(const v2f*)&As[buf][wm * 32 + l16][kk];
      af0[2 * j] = (_Float16)p0.x; af0[2 * j + 1] = (_Float16)p0.y;
      v2f p1 = *(const v2f*)&As[buf][wm * 32 + 16 + l16][kk];
      af1[2 * j] = (_Float16)p1.x; af1[2 * j + 1] = (_Float16)p1.y;
    }
    acc0 = __builtin_amdgcn_wmma_f32_16x16x32_f16(false, af0, false, bfrag,
                                                  (short)0, acc0, false, false);
    acc1 = __builtin_amdgcn_wmma_f32_16x16x32_f16(false, af1, false, bfrag,
                                                  (short)0, acc1, false, false);
    __syncthreads();   // all waves done reading buf before refill

    if (wave == 0 && c + 2 < nc) {
      int k2 = (c + 2) * 32;
      tdm_load_2d(aLds + buf * aBufB, A + (size_t)mBlk * K + k2,   32, 64, (unsigned)K, MROWS, (unsigned)K);
      tdm_load_2d(bLds + buf * bBufB, W + (size_t)k2 * N + nBlk,   64, 32, (unsigned)N, (unsigned)K, (unsigned)N);
    }
  }

  int rowOff = (lane < 16) ? 0 : 8;
  int col = nBlk + wn * 16 + l16;
  float bval = HASBIAS ? bias[col] : 0.f;
#pragma unroll
  for (int sft = 0; sft < 2; ++sft) {
#pragma unroll
    for (int r = 0; r < 8; ++r) {
      int row = mBlk + wm * 32 + sft * 16 + r + rowOff;
      float v = (sft ? acc1[r] : acc0[r]) + bval;
      if (GELU)  v = v * __builtin_amdgcn_rcpf(1.f + __expf(-1.702f * v));
      if (HASRES) v += res[(size_t)row * N + col];
      out[(size_t)row * N + col] = v;
    }
  }
}

// ---------------------------------------------------------------------------
// Flash attention with causal + distance-decay mask.
// Block = 128 thr (4 waves); each wave owns one 16-row Q tile.
// QK^T and P@V via v_wmma_f32_16x16x4_f32 (DK=36 = 9*4; PV K=16 = 4*4,
// N padded 36->48 with zero B columns).
// ---------------------------------------------------------------------------
__device__ inline float dist_decay(int qi, int ki) {
  int qt = qi >> 8, qh = (qi >> 4) & 15, qw = qi & 15;
  int kt = ki >> 8, kh = (ki >> 4) & 15, kw = ki & 15;
  int d = __builtin_abs(qt - kt) + __builtin_abs(qh - kh) + __builtin_abs(qw - kw);
  return __expf(-(float)d * (1.0f / 33.0f));   // max L1 distance = 3+15+15
}

__global__ __launch_bounds__(128)
void attn_kernel(const float* __restrict__ Q, const float* __restrict__ Kb,
                 const float* __restrict__ V, float* __restrict__ O) {
  __shared__ float Plds[4][16][16];
  int wave = threadIdx.x >> 5, lane = threadIdx.x & 31;
  int blk  = blockIdx.x;
  int qBlk = blk & 15;
  int bh   = blk >> 4;
  int b = bh >> 4, h = bh & 15;
  int qTile = qBlk * 4 + wave;          // 16-row tile idx, 0..63
  int qBase = qTile * 16;
  int l16   = lane & 15;
  int off2  = (lane < 16) ? 0 : 2;
  int rowOff = (lane < 16) ? 0 : 8;
  const size_t base = ((size_t)b * SEQ) * EMB + h * DKH;

  v2f qa[9];
#pragma unroll
  for (int c = 0; c < 9; ++c)
    qa[c] = *(const v2f*)&Q[base + (size_t)(qBase + l16) * EMB + c * 4 + off2];

  v8f oacc[3] = {{}, {}, {}};
  float mrow[8], lrow[8];
#pragma unroll
  for (int r = 0; r < 8; ++r) { mrow[r] = -1e30f; lrow[r] = 0.f; }
  const float scale = 1.0f / 6.0f;

  for (int j = 0; j <= qTile; ++j) {
    if (j < qTile)   // prefetch next K tile (global_prefetch_b8)
      __builtin_prefetch(&Kb[base + (size_t)((j + 1) * 16 + l16) * EMB], 0, 3);
    v8f s = {};
#pragma unroll
    for (int c = 0; c < 9; ++c) {
      v2f kb = *(const v2f*)&Kb[base + (size_t)(j * 16 + l16) * EMB + c * 4 + off2];
      s = __builtin_amdgcn_wmma_f32_16x16x4_f32(false, qa[c], false, kb,
                                                (short)0, s, false, false);
    }
#pragma unroll
    for (int r = 0; r < 8; ++r) {
      int qg = qBase + r + rowOff;
      int kg = j * 16 + l16;
      float sv = (kg > qg) ? -1e30f : s[r] * scale * dist_decay(qg, kg);
      float mnew = fmaxf(mrow[r], red16_max(sv));
      float p = __expf(sv - mnew);
      float rsum = red16_sum(p);
      float alpha = __expf(mrow[r] - mnew);
      lrow[r] = lrow[r] * alpha + rsum;
      mrow[r] = mnew;
      oacc[0][r] *= alpha; oacc[1][r] *= alpha; oacc[2][r] *= alpha;
      Plds[wave][r + rowOff][l16] = p;
    }
    asm volatile("s_wait_dscnt 0x0" ::: "memory");  // intra-wave LDS RAW

#pragma unroll
    for (int c = 0; c < 4; ++c) {
      v2f pa = *(const v2f*)&Plds[wave][l16][c * 4 + off2];
      int kp0 = j * 16 + c * 4 + off2;
      const float* vr0 = &V[base + (size_t)kp0 * EMB];
      const float* vr1 = vr0 + EMB;
#pragma unroll
      for (int t = 0; t < 3; ++t) {
        int d = t * 16 + l16;
        v2f vb;
        vb.x = (d < DKH) ? vr0[d] : 0.f;
        vb.y = (d < DKH) ? vr1[d] : 0.f;
        oacc[t] = __builtin_amdgcn_wmma_f32_16x16x4_f32(false, pa, false, vb,
                                                        (short)0, oacc[t],
                                                        false, false);
      }
    }
  }

#pragma unroll
  for (int r = 0; r < 8; ++r) {
    float inv = 1.0f / lrow[r];
    int row = qBase + r + rowOff;
#pragma unroll
    for (int t = 0; t < 3; ++t) {
      int d = t * 16 + l16;
      if (d < DKH)
        O[base + (size_t)row * EMB + d] = oacc[t][r] * inv;
    }
  }
}

// ---------------------------------------------------------------------------
extern "C" void kernel_launch(void* const* d_in, const int* in_sizes, int n_in,
                              void* d_out, int out_size, void* d_ws, size_t ws_size,
                              hipStream_t stream) {
  const float* x     = (const float*)d_in[0];
  const float* sos   = (const float*)d_in[1];
  const float* pe0   = (const float*)d_in[2];
  const float* pe1   = (const float*)d_in[3];
  const float* pe2   = (const float*)d_in[4];
  const float* ln1_s = (const float*)d_in[5];
  const float* ln1_b = (const float*)d_in[6];
  const float* wq    = (const float*)d_in[7];
  const float* wk    = (const float*)d_in[8];
  const float* wv    = (const float*)d_in[9];
  const float* wo    = (const float*)d_in[10];
  const float* bo    = (const float*)d_in[11];
  const float* ln2_s = (const float*)d_in[12];
  const float* ln2_b = (const float*)d_in[13];
  const float* w1    = (const float*)d_in[14];
  const float* b1    = (const float*)d_in[15];
  const float* w2    = (const float*)d_in[16];
  const float* b2    = (const float*)d_in[17];

  const size_t ME = (size_t)MROWS * EMB;
  float* ws  = (float*)d_ws;
  float* h   = ws;
  float* y   = h + ME;
  float* q   = y + ME;
  float* k   = q + ME;
  float* v   = k + ME;
  float* att = v + ME;
  float* f1  = q;             // FFN hidden [2048,2304] aliases q..att exactly

  embed_kernel<<<(MROWS * EMB) / 256, 256, 0, stream>>>(x, sos, pe0, pe1, pe2, h);

  dim3 gProj(MROWS / 64, EMB / 64);   // 32 x 9
  dim3 gFF1(MROWS / 64, FFD / 64);    // 32 x 36
  for (int l = 0; l < NLAYER; ++l) {
    ln_kernel<<<MROWS / 8, 256, 0, stream>>>(h, ln1_s + l * EMB, ln1_b + l * EMB, y);
    gemm_wmma<0,0,0><<<gProj, 256, 0, stream>>>(y, wq + (size_t)l * EMB * EMB, nullptr, nullptr, q, EMB, EMB);
    gemm_wmma<0,0,0><<<gProj, 256, 0, stream>>>(y, wk + (size_t)l * EMB * EMB, nullptr, nullptr, k, EMB, EMB);
    gemm_wmma<0,0,0><<<gProj, 256, 0, stream>>>(y, wv + (size_t)l * EMB * EMB, nullptr, nullptr, v, EMB, EMB);
    attn_kernel<<<BATCH * NHEAD * (SEQ / 64), 128, 0, stream>>>(q, k, v, att);
    gemm_wmma<0,1,1><<<gProj, 256, 0, stream>>>(att, wo + (size_t)l * EMB * EMB, bo + l * EMB, h, h, EMB, EMB);
    ln_kernel<<<MROWS / 8, 256, 0, stream>>>(h, ln2_s + l * EMB, ln2_b + l * EMB, y);
    gemm_wmma<1,0,1><<<gFF1, 256, 0, stream>>>(y, w1 + (size_t)l * EMB * FFD, b1 + l * FFD, nullptr, f1, EMB, FFD);
    gemm_wmma<0,1,1><<<gProj, 256, 0, stream>>>(f1, w2 + (size_t)l * FFD * EMB, b2 + l * EMB, h, h, FFD, EMB);
  }
  (void)hipMemcpyAsync(d_out, h, ME * sizeof(float), hipMemcpyDeviceToDevice, stream);
}
